// ConditionalDDiTBlock_39075612459670
// MI455X (gfx1250) — matrix operand order repaired and checked by reading, not compiled, not run on hardware
//
#include <hip/hip_runtime.h>
#include <hip/hip_bf16.h>

// ---------------------------------------------------------------------------
// ConditionalDDiTBlock for MI455X (gfx1250, wave32, WMMA + TDM).
// All heavy math in bf16 WMMA (v_wmma_f32_16x16x32_bf16), fp32 accumulate.
// GEMM weight tiles staged in LDS via the Tensor Data Mover (tensor_load_to_lds,
// TENSORcnt double-buffer pipeline); activations streamed from global (L2-resident).
// ---------------------------------------------------------------------------

typedef __attribute__((ext_vector_type(16))) __bf16 v16bf;
typedef __attribute__((ext_vector_type(8)))  float  v8f;

constexpr int B_    = 2;
constexpr int N_    = 1024;
constexpr int S_    = 2048;   // 2*N
constexpr int D_    = 1024;
constexpr int H_    = 16;
constexpr int HD_   = 64;
constexpr int TV_   = 256;
constexpr int COND_ = 1024;

typedef __hip_bfloat16 bf;

#if defined(__has_builtin)
#  if __has_builtin(__builtin_amdgcn_tensor_load_to_lds) && __has_builtin(__builtin_amdgcn_s_wait_tensorcnt)
#    define USE_TDM 1
#  else
#    define USE_TDM 0
#  endif
#else
#  define USE_TDM 0
#endif

__device__ __forceinline__ v8f wmma_bf16(v16bf a, v16bf b, v8f c) {
  // (neg_a, A, neg_b, B, c_mod, C, reuse_a, reuse_b)
  return __builtin_amdgcn_wmma_f32_16x16x32_bf16(false, a, false, b, (short)0, c,
                                                 false, false);
}

union FragU { uint4 q[2]; v16bf v; };

// A fragment (16x32, row-major, K contiguous). lane r = row, hi = lane>>4.
// Lanes 0-15 hold K = k0+0..15, lanes 16-31 K = k0+16..31 (contiguous, matches B).
__device__ __forceinline__ v16bf load_frag_a(const bf* __restrict__ p, int ld,
                                             int row0, int k0, int r, int hi) {
  const bf* b = p + (size_t)(row0 + r) * ld + k0 + hi * 16;
  FragU u;
  u.q[0] = *(const uint4*)(b);
  u.q[1] = *(const uint4*)(b + 8);
  return u.v;
}

// B fragment (32x16) for NT gemm from global: W row-major (N,K); B[k][n]=W[n][k].
// Same contiguous k order as A: lane n holds K = k0 + hi*16 + [0..15] contiguous.
__device__ __forceinline__ v16bf load_frag_b(const bf* __restrict__ p, int ld,
                                             int n0, int k0, int r, int hi) {
  const bf* b = p + (size_t)(n0 + r) * ld + k0 + hi * 16;
  FragU u;
  u.q[0] = *(const uint4*)(b);
  u.q[1] = *(const uint4*)(b + 8);
  return u.v;
}

// Same B fragment out of an LDS-staged 128x32 row-major tile (ld = 32).
__device__ __forceinline__ v16bf frag_b_lds(const bf* p) {
  FragU u;
  u.q[0] = *(const uint4*)(p);
  u.q[1] = *(const uint4*)(p + 8);
  return u.v;
}

__device__ __forceinline__ float gelu_tanh(float v) {
  float c = 0.7978845608028654f * (v + 0.044715f * v * v * v);
  return 0.5f * v * (1.0f + tanhf(c));
}

#if USE_TDM
typedef __attribute__((ext_vector_type(4))) unsigned int u32x4;
typedef __attribute__((ext_vector_type(8))) int i32x8;
typedef __attribute__((ext_vector_type(4))) int i32x4;

// Issue a TDM load of a 128-row x 32-element bf16 tile (row stride = ld elems)
// from gsrc into LDS at byte offset lds_off (rows packed contiguously, 64B/row).
// D# built per CDNA5 ISA (group0: count=1, lds, global,
// type=2; group1: data_size=2B, tensor 32x128, tile 32x128, stride0=ld).
// 6-arg builtin variant (clang-23 / therock-10.0 headers).
__device__ __forceinline__ void tdm_load_tile_128x32(const bf* gsrc, int ld,
                                                     unsigned lds_off) {
  unsigned long long ga = (unsigned long long)(size_t)gsrc;
  u32x4 g0;
  g0.x = 1u;                                      // count=1, user mode
  g0.y = lds_off;                                 // lds_addr (bytes)
  g0.z = (unsigned)(ga & 0xFFFFFFFFu);            // global_addr[31:0]
  g0.w = (unsigned)((ga >> 32) & 0x01FFFFFFu) | (2u << 30);  // addr[56:32] | type=2
  i32x8 g1;
  g1[0] = 1 << 16;                                // wg_mask=0, data_size=1 (2B)
  g1[1] = (int)(32u << 16);                       // tensor_dim0 = 32
  g1[2] = (int)(128u << 16);                      // tensor_dim1 = 128
  g1[3] = (int)(32u << 16);                       // tile_dim0 = 32
  g1[4] = 128;                                    // tile_dim1 = 128, tile_dim2 = 0
  g1[5] = ld;                                     // tensor_dim0_stride (elems)
  g1[6] = 0;
  g1[7] = 0;
  i32x4 z4 = {0, 0, 0, 0};                        // groups 2/3 unused (2D)
  i32x8 z8 = {0, 0, 0, 0, 0, 0, 0, 0};            // trailing group (clang-23 form)
  __builtin_amdgcn_tensor_load_to_lds(g0, g1, z4, z4, z8, 0);
}
#endif

// ---------------------------------------------------------------------------
// f32 -> bf16 convert
// ---------------------------------------------------------------------------
__global__ __launch_bounds__(256)
void f32_to_bf16_kernel(const float* __restrict__ in, bf* __restrict__ out, int n) {
  int i = blockIdx.x * 256 + threadIdx.x;
  if (i < n) out[i] = __float2bfloat16(in[i]);
}

// ---------------------------------------------------------------------------
// ada = c @ W_ada^T + b_ada   (B x 6D), one wave per output
// ---------------------------------------------------------------------------
__global__ __launch_bounds__(256)
void ada_kernel(const float* __restrict__ c, const float* __restrict__ Wa,
                const float* __restrict__ ba, float* __restrict__ ada) {
  int gw   = (blockIdx.x * 256 + threadIdx.x) >> 5;
  int lane = threadIdx.x & 31;
  int total = B_ * 6 * D_;
  if (gw >= total) return;
  int b = gw / (6 * D_);
  int j = gw % (6 * D_);
  const float* cv = c + (size_t)b * COND_;
  const float* wv = Wa + (size_t)j * COND_;
  float s = 0.f;
  for (int k = lane; k < COND_; k += 32) s += cv[k] * wv[k];
  #pragma unroll
  for (int off = 16; off > 0; off >>= 1) s += __shfl_xor(s, off, 32);
  if (lane == 0) ada[(size_t)b * 6 * D_ + j] = s + ba[j];
}

// ---------------------------------------------------------------------------
// LayerNorm (+ optional adaLN scale/shift) -> bf16. One block per row (D=1024).
// sc_off < 0  => plain LN.
// ---------------------------------------------------------------------------
__global__ __launch_bounds__(256)
void ln_mod_kernel(const float* __restrict__ x, const float* __restrict__ w,
                   const float* __restrict__ ada, int sc_off, int sh_off,
                   bf* __restrict__ out) {
  __shared__ float red[256];
  int row = blockIdx.x;
  int b   = row / S_;
  int tid = threadIdx.x;
  const float* xr = x + (size_t)row * D_;
  float v[4];
  float s = 0.f;
  #pragma unroll
  for (int i = 0; i < 4; i++) { v[i] = xr[tid + i * 256]; s += v[i]; }
  red[tid] = s; __syncthreads();
  for (int st = 128; st > 0; st >>= 1) { if (tid < st) red[tid] += red[tid + st]; __syncthreads(); }
  float mu = red[0] * (1.0f / D_);
  __syncthreads();
  s = 0.f;
  #pragma unroll
  for (int i = 0; i < 4; i++) { float d = v[i] - mu; s += d * d; }
  red[tid] = s; __syncthreads();
  for (int st = 128; st > 0; st >>= 1) { if (tid < st) red[tid] += red[tid + st]; __syncthreads(); }
  float inv = rsqrtf(red[0] * (1.0f / D_) + 1e-5f);
  const float* ab = ada + (size_t)b * 6 * D_;
  #pragma unroll
  for (int i = 0; i < 4; i++) {
    int d = tid + i * 256;
    float y = (v[i] - mu) * inv * w[d];
    if (sc_off >= 0) y = y * (1.0f + ab[sc_off + d]) + ab[sh_off + d];
    out[(size_t)row * D_ + d] = __float2bfloat16(y);
  }
}

// ---------------------------------------------------------------------------
// bf16 NT GEMM: Out(MxN) = A(MxK) @ W(NxK)^T, fused epilogue.
// Block 256 threads = 8 waves; wave tile 32x64; block tile 128x128.
// Weight tile (128x32 bf16, 8KB) staged in LDS via TDM, double-buffered:
//   wave0 issues tensor_load_to_lds for k+32, all waves compute k from LDS,
//   wave0 s_wait_tensorcnt(0), block barrier, swap buffers.
// Requires M%128==0, N%128==0, K%32==0 (true for all uses here).
// ---------------------------------------------------------------------------
__global__ __launch_bounds__(256)
void gemm_bf16_nt(const bf* __restrict__ A, int lda,
                  const bf* __restrict__ W, int ldw,
                  void* __restrict__ Out, int ldo, int out_bf16,
                  int K,
                  const float* __restrict__ bias, int do_gelu,
                  const float* __restrict__ gate, int gstride, int rpb,
                  const float* __restrict__ resid, int ldr) {
  __shared__ bf bsm[2][128 * 32];                 // 2 x 8KB weight tiles

  int lane = threadIdx.x & 31;
  int wave = threadIdx.x >> 5;
  int r  = lane & 15;
  int hi = lane >> 4;
  int m0  = blockIdx.y * 128 + (wave >> 1) * 32;
  int nb  = blockIdx.x * 128;                     // block N origin
  int wn0 = (wave & 1) * 64;                      // wave N offset inside tile

  v8f acc[2][4] = {};

#if USE_TDM
  unsigned lds_base = (unsigned)(size_t)(&bsm[0][0]);
  // prologue: stage k0 = 0 into buffer 0
  if (wave == 0) {
    tdm_load_tile_128x32(W + (size_t)nb * ldw, ldw, lds_base);
    __builtin_amdgcn_s_wait_tensorcnt(0);
  }
  __syncthreads();
#else
  {
    int rr = threadIdx.x >> 1, hh = (threadIdx.x & 1) * 16;
    const bf* src = W + (size_t)(nb + rr) * ldw + hh;
    bf* dst = &bsm[0][rr * 32 + hh];
    *(uint4*)(dst)     = *(const uint4*)(src);
    *(uint4*)(dst + 8) = *(const uint4*)(src + 8);
  }
  __syncthreads();
#endif

  for (int k0 = 0; k0 < K; k0 += 32) {
    int cur = (k0 >> 5) & 1;
    int nxt = cur ^ 1;
    bool more = (k0 + 32) < K;

#if USE_TDM
    if (wave == 0 && more)
      tdm_load_tile_128x32(W + (size_t)nb * ldw + (k0 + 32), ldw,
                           lds_base + (unsigned)nxt * (128 * 32 * 2));
#else
    if (more) {
      int rr = threadIdx.x >> 1, hh = (threadIdx.x & 1) * 16;
      const bf* src = W + (size_t)(nb + rr) * ldw + (k0 + 32) + hh;
      bf* dst = &bsm[nxt][rr * 32 + hh];
      *(uint4*)(dst)     = *(const uint4*)(src);
      *(uint4*)(dst + 8) = *(const uint4*)(src + 8);
    }
#endif

    // software prefetch of the A stream (global_prefetch_b8)
    if (k0 + 64 < K) __builtin_prefetch(A + (size_t)(m0 + r) * lda + k0 + 64, 0, 0);

    v16bf a0 = load_frag_a(A, lda, m0,      k0, r, hi);
    v16bf a1 = load_frag_a(A, lda, m0 + 16, k0, r, hi);
    const bf* bt = &bsm[cur][0];
    v16bf b0 = frag_b_lds(bt + (wn0 +  0 + r) * 32 + hi * 16);
    v16bf b1 = frag_b_lds(bt + (wn0 + 16 + r) * 32 + hi * 16);
    v16bf b2 = frag_b_lds(bt + (wn0 + 32 + r) * 32 + hi * 16);
    v16bf b3 = frag_b_lds(bt + (wn0 + 48 + r) * 32 + hi * 16);
    acc[0][0] = wmma_bf16(a0, b0, acc[0][0]);
    acc[0][1] = wmma_bf16(a0, b1, acc[0][1]);
    acc[0][2] = wmma_bf16(a0, b2, acc[0][2]);
    acc[0][3] = wmma_bf16(a0, b3, acc[0][3]);
    acc[1][0] = wmma_bf16(a1, b0, acc[1][0]);
    acc[1][1] = wmma_bf16(a1, b1, acc[1][1]);
    acc[1][2] = wmma_bf16(a1, b2, acc[1][2]);
    acc[1][3] = wmma_bf16(a1, b3, acc[1][3]);

#if USE_TDM
    if (wave == 0 && more) __builtin_amdgcn_s_wait_tensorcnt(0);
#endif
    __syncthreads();
  }

  #pragma unroll
  for (int mi = 0; mi < 2; mi++) {
    #pragma unroll
    for (int ni = 0; ni < 4; ni++) {
      #pragma unroll
      for (int i = 0; i < 8; i++) {
        int row = m0 + mi * 16 + hi * 8 + i;
        int col = nb + wn0 + ni * 16 + r;
        float v = acc[mi][ni][i];
        if (bias)    v += bias[col];
        if (do_gelu) v  = gelu_tanh(v);
        if (gate)    v *= gate[(size_t)(row / rpb) * gstride + col];
        if (resid)   v += resid[(size_t)row * ldr + col];
        if (out_bf16)
          ((bf*)Out)[(size_t)row * ldo + col] = __float2bfloat16(v);
        else
          ((float*)Out)[(size_t)row * ldo + col] = v;
      }
    }
  }
}

// ---------------------------------------------------------------------------
// RoPE applied in-place on bf16 qkv (channels q,k), plus V -> transposed copy.
// grid: B*S blocks, 512 threads = (h, pair-index d2).
// ---------------------------------------------------------------------------
__global__ __launch_bounds__(512)
void rotary_pack(bf* __restrict__ qkv, const float* __restrict__ cosb,
                 const float* __restrict__ sinb, bf* __restrict__ vT) {
  int row = blockIdx.x;              // b*S + s
  int b = row / S_, s = row % S_;
  int pos = s % N_;
  int h  = threadIdx.x >> 5;
  int d2 = threadIdx.x & 31;
  bf* base = qkv + (size_t)row * 3 * D_ + h * HD_;
  #pragma unroll
  for (int ch = 0; ch < 2; ch++) {   // q, k
    bf* pch = base + ch * D_;
    size_t ci = ((size_t)pos * 3 + ch) * HD_;
    float c1 = cosb[ci + d2],      s1 = sinb[ci + d2];
    float c2 = cosb[ci + d2 + 32], s2 = sinb[ci + d2 + 32];
    float u1 = __bfloat162float(pch[d2]);
    float u2 = __bfloat162float(pch[d2 + 32]);
    pch[d2]      = __float2bfloat16(u1 * c1 - u2 * s1);
    pch[d2 + 32] = __float2bfloat16(u2 * c2 + u1 * s2);
  }
  const bf* pv = base + 2 * D_;
  size_t vbase = (size_t)(b * H_ + h) * HD_ * (size_t)S_;
  vT[vbase + (size_t)d2 * S_ + s]        = pv[d2];
  vT[vbase + (size_t)(d2 + 32) * S_ + s] = pv[d2 + 32];
}

// ---------------------------------------------------------------------------
// Cross-attn V transpose: kv (B*TV, 2D) col D+h*64+hd -> vt (B,H,HD,TV)
// ---------------------------------------------------------------------------
__global__ __launch_bounds__(1024)
void pack_vt2(const bf* __restrict__ kv, bf* __restrict__ vt) {
  int row = blockIdx.x;              // b*TV + t
  int b = row / TV_, t = row % TV_;
  int h  = threadIdx.x >> 6;
  int hd = threadIdx.x & 63;
  vt[((size_t)(b * H_ + h) * HD_ + hd) * TV_ + t] =
      kv[(size_t)row * 2 * D_ + D_ + h * HD_ + hd];
}

// ---------------------------------------------------------------------------
// Attention: one wave per 16-query tile of one (b,h). Scores for all SK keys
// staged in LDS (16*SK floats), exact softmax, then P@V via WMMA.
// MASK2D=1: mask is (S,SK) bytes indexed [q,key]; else (B,SK) bytes [b,key].
// ---------------------------------------------------------------------------
template <int SK, int MASK2D>
__global__ void attn_kernel(const bf* __restrict__ Qp, size_t qbs, int ldq,
                            const bf* __restrict__ Kp, size_t kbs, int ldk,
                            const bf* __restrict__ VTp,
                            const unsigned char* __restrict__ mask,
                            bf* __restrict__ O) {
  extern __shared__ float smem[];    // 16 * SK floats
  const float scale = 0.125f;        // 1/sqrt(HD=64)
  int lane = threadIdx.x;
  int r  = lane & 15;
  int hi = lane >> 4;
  int q0 = blockIdx.x * 16;
  int bh = blockIdx.y;
  int b = bh / H_, h = bh % H_;
  const bf* Qb = Qp + (size_t)b * qbs + h * HD_;
  const bf* Kb = Kp + (size_t)b * kbs + h * HD_;
  const bf* Vb = VTp + (size_t)bh * HD_ * SK;

  v16bf qa0 = load_frag_a(Qb, ldq, q0, 0,  r, hi);
  v16bf qa1 = load_frag_a(Qb, ldq, q0, 32, r, hi);

  // Phase A: scores
  for (int kt = 0; kt < SK; kt += 16) {
    v16bf kb0 = load_frag_b(Kb, ldk, kt, 0,  r, hi);
    v16bf kb1 = load_frag_b(Kb, ldk, kt, 32, r, hi);
    v8f sacc = {};
    sacc = wmma_bf16(qa0, kb0, sacc);
    sacc = wmma_bf16(qa1, kb1, sacc);
    int key = kt + r;
    #pragma unroll
    for (int i = 0; i < 8; i++) {
      int m = hi * 8 + i;
      float v = sacc[i] * scale;
      bool ok = MASK2D ? (mask[(size_t)(q0 + m) * SK + key] != 0)
                       : (mask[(size_t)b * SK + key] != 0);
      smem[m * SK + key] = ok ? v : -1e30f;
    }
  }
  __syncthreads();

  // Phase B: softmax, 2 lanes per row
  {
    float* p = smem + r * SK + hi * (SK / 2);
    float mx = -3.0e38f;
    for (int j = 0; j < SK / 2; j++) mx = fmaxf(mx, p[j]);
    mx = fmaxf(mx, __shfl_xor(mx, 16, 32));
    float sum = 0.f;
    for (int j = 0; j < SK / 2; j++) sum += __expf(p[j] - mx);
    sum += __shfl_xor(sum, 16, 32);
    float inv = 1.0f / sum;
    for (int j = 0; j < SK / 2; j++) p[j] = __expf(p[j] - mx) * inv;
  }
  __syncthreads();

  // Phase C: O = P @ V  (n = HD = 64 -> 4 tiles)
  v8f o[4] = {};
  for (int kt = 0; kt < SK; kt += 32) {
    const float* prow = smem + r * SK + kt + hi * 16;
    v16bf pa;
    #pragma unroll
    for (int j = 0; j < 8; j++) {
      pa[j]     = (__bf16)prow[j];
      pa[8 + j] = (__bf16)prow[8 + j];
    }
    #pragma unroll
    for (int ni = 0; ni < 4; ni++) {
      v16bf vbf = load_frag_b(Vb, SK, ni * 16, kt, r, hi);
      o[ni] = wmma_bf16(pa, vbf, o[ni]);
    }
  }
  #pragma unroll
  for (int ni = 0; ni < 4; ni++) {
    #pragma unroll
    for (int i = 0; i < 8; i++) {
      size_t row = (size_t)b * S_ + q0 + hi * 8 + i;
      O[row * D_ + h * HD_ + ni * 16 + r] = __float2bfloat16(o[ni][i]);
    }
  }
}

// ---------------------------------------------------------------------------
// Launch
// ---------------------------------------------------------------------------
extern "C" void kernel_launch(void* const* d_in, const int* in_sizes, int n_in,
                              void* d_out, int out_size, void* d_ws, size_t ws_size,
                              hipStream_t stream) {
  (void)in_sizes; (void)n_in; (void)out_size; (void)ws_size;
  const float* x       = (const float*)d_in[0];
  const float* c       = (const float*)d_in[1];
  const float* enc     = (const float*)d_in[2];
  const unsigned char* enc_mask = (const unsigned char*)d_in[3];
  const unsigned char* mask     = (const unsigned char*)d_in[4];
  const float* cosb    = (const float*)d_in[5];
  const float* sinb    = (const float*)d_in[6];
  const float* ln1_w   = (const float*)d_in[7];
  const float* W_qkv   = (const float*)d_in[8];
  const float* W_ao    = (const float*)d_in[9];
  const float* ca_ln   = (const float*)d_in[10];
  const float* ca_Wq   = (const float*)d_in[11];
  const float* ca_Wkv  = (const float*)d_in[12];
  const float* ca_Wo   = (const float*)d_in[13];
  const float* ln2_w   = (const float*)d_in[14];
  const float* W_m1    = (const float*)d_in[15];
  const float* b_m1    = (const float*)d_in[16];
  const float* W_m2    = (const float*)d_in[17];
  const float* b_m2    = (const float*)d_in[18];
  const float* W_adaP  = (const float*)d_in[19];
  const float* b_adaP  = (const float*)d_in[20];

  size_t off = 0;
  auto carve = [&](size_t bytes) -> char* {
    char* p = (char*)d_ws + off;
    off += (bytes + 255) & ~(size_t)255;
    return p;
  };

  bf*    wb_qkv  = (bf*)carve((size_t)3 * D_ * D_ * 2);
  bf*    wb_ao   = (bf*)carve((size_t)D_ * D_ * 2);
  bf*    wb_caq  = (bf*)carve((size_t)D_ * D_ * 2);
  bf*    wb_cakv = (bf*)carve((size_t)2 * D_ * D_ * 2);
  bf*    wb_cao  = (bf*)carve((size_t)D_ * D_ * 2);
  bf*    wb_m1   = (bf*)carve((size_t)4 * D_ * D_ * 2);
  bf*    wb_m2   = (bf*)carve((size_t)4 * D_ * D_ * 2);
  bf*    enc_bf  = (bf*)carve((size_t)B_ * TV_ * D_ * 2);
  float* ada     = (float*)carve((size_t)B_ * 6 * D_ * 4);
  bf*    xn_bf   = (bf*)carve((size_t)B_ * S_ * D_ * 2);     // reused for all LN outputs
  bf*    qkv_bf  = (bf*)carve((size_t)B_ * S_ * 3 * D_ * 2);
  bf*    vT      = (bf*)carve((size_t)B_ * H_ * HD_ * S_ * 2);
  bf*    o_bf    = (bf*)carve((size_t)B_ * S_ * D_ * 2);     // attn outputs (self & cross)
  float* x1      = (float*)carve((size_t)B_ * S_ * D_ * 4);
  bf*    qc_bf   = (bf*)carve((size_t)B_ * S_ * D_ * 2);
  bf*    kv_bf   = (bf*)carve((size_t)B_ * TV_ * 2 * D_ * 2);
  bf*    vT2     = (bf*)carve((size_t)B_ * H_ * HD_ * TV_ * 2);
  float* x2      = (float*)carve((size_t)B_ * S_ * D_ * 4);
  bf*    h1_bf   = (bf*)carve((size_t)B_ * S_ * 4 * D_ * 2);

  dim3 blk256(256);
  auto cvt = [&](const float* src, bf* dst, int n) {
    f32_to_bf16_kernel<<<dim3((n + 255) / 256), blk256, 0, stream>>>(src, dst, n);
  };
  cvt(W_qkv,  wb_qkv,  3 * D_ * D_);
  cvt(W_ao,   wb_ao,   D_ * D_);
  cvt(ca_Wq,  wb_caq,  D_ * D_);
  cvt(ca_Wkv, wb_cakv, 2 * D_ * D_);
  cvt(ca_Wo,  wb_cao,  D_ * D_);
  cvt(W_m1,   wb_m1,   4 * D_ * D_);
  cvt(W_m2,   wb_m2,   4 * D_ * D_);
  cvt(enc,    enc_bf,  B_ * TV_ * D_);

  // adaLN vector
  {
    int waves = B_ * 6 * D_;
    ada_kernel<<<dim3((waves * 32 + 255) / 256), blk256, 0, stream>>>(c, W_adaP, b_adaP, ada);
  }

  auto gemm = [&](const bf* A, int lda, const bf* W, int ldw, void* Out, int ldo,
                  int obf, int M, int Ncols, int K, const float* bias, int gelu,
                  const float* gate, const float* resid) {
    gemm_bf16_nt<<<dim3(Ncols / 128, M / 128), blk256, 0, stream>>>(
        A, lda, W, ldw, Out, ldo, obf, K, bias, gelu, gate, 6 * D_, S_, resid, D_);
  };

  // --- self-attention branch ---
  ln_mod_kernel<<<dim3(B_ * S_), blk256, 0, stream>>>(x, ln1_w, ada, D_, 0, xn_bf);
  gemm(xn_bf, D_, wb_qkv, D_, qkv_bf, 3 * D_, 1, B_ * S_, 3 * D_, D_,
       nullptr, 0, nullptr, nullptr);
  rotary_pack<<<dim3(B_ * S_), dim3(512), 0, stream>>>(qkv_bf, cosb, sinb, vT);
  attn_kernel<S_, 1><<<dim3(S_ / 16, B_ * H_), dim3(32), 16 * S_ * 4, stream>>>(
      qkv_bf,        (size_t)S_ * 3 * D_, 3 * D_,
      qkv_bf + D_,   (size_t)S_ * 3 * D_, 3 * D_,
      vT, mask, o_bf);
  // x1 = g_msa * (o @ W_ao^T) + x
  gemm(o_bf, D_, wb_ao, D_, x1, D_, 0, B_ * S_, D_, D_,
       nullptr, 0, ada + 2 * D_, x);

  // --- cross-attention branch ---
  ln_mod_kernel<<<dim3(B_ * S_), blk256, 0, stream>>>(x1, ca_ln, ada, -1, -1, xn_bf);
  gemm(xn_bf, D_, wb_caq, D_, qc_bf, D_, 1, B_ * S_, D_, D_,
       nullptr, 0, nullptr, nullptr);
  gemm(enc_bf, D_, wb_cakv, D_, kv_bf, 2 * D_, 1, B_ * TV_, 2 * D_, D_,
       nullptr, 0, nullptr, nullptr);
  pack_vt2<<<dim3(B_ * TV_), dim3(1024), 0, stream>>>(kv_bf, vT2);
  attn_kernel<TV_, 0><<<dim3(S_ / 16, B_ * H_), dim3(32), 16 * TV_ * 4, stream>>>(
      qc_bf, (size_t)S_ * D_, D_,
      kv_bf, (size_t)TV_ * 2 * D_, 2 * D_,
      vT2, enc_mask, o_bf);
  // x2 = x1 + o @ W_cao^T
  gemm(o_bf, D_, wb_cao, D_, x2, D_, 0, B_ * S_, D_, D_,
       nullptr, 0, nullptr, x1);

  // --- MLP branch ---
  ln_mod_kernel<<<dim3(B_ * S_), blk256, 0, stream>>>(x2, ln2_w, ada, 4 * D_, 3 * D_, xn_bf);
  gemm(xn_bf, D_, wb_m1, D_, h1_bf, 4 * D_, 1, B_ * S_, 4 * D_, D_,
       b_m1, 1, nullptr, nullptr);
  // out = g_mlp * (h1 @ W_m2^T + b_m2) + x2
  gemm(h1_bf, 4 * D_, wb_m2, 4 * D_, d_out, D_, 0, B_ * S_, D_, 4 * D_,
       b_m2, 0, ada + 5 * D_, x2);
}
